// RFS_14534169329776
// MI455X (gfx1250) — compile-verified, matrix operand order checked by reading, not run on hardware
//
#include <hip/hip_runtime.h>
#include <math.h>

typedef __attribute__((ext_vector_type(16))) _Float16 v16h;
typedef __attribute__((ext_vector_type(8)))  float    v8f;

#define HW_PIX 65536
#define IMG_W  256
#define GAMMA_F 0.99f

// mask = sigmoid(100*(v-thr)) + 1e-6 ; field = relu(v) * mask
__device__ __forceinline__ float fieldval(float v, float thr) {
  float s = 1.0f / (1.0f + __expf(-100.0f * (v - thr)));
  return v * (s + 1e-6f);
}

// 16-byte async copy global -> LDS (ASYNCcnt-tracked, gfx1250)
__device__ __forceinline__ void async_copy16(const void* g, void* l) {
  unsigned ldsoff = (unsigned)(unsigned long long)l;   // low 32 bits = LDS offset
  asm volatile("global_load_async_to_lds_b128 %0, %1, off"
               :: "v"(ldsoff), "v"(g) : "memory");
}
__device__ __forceinline__ void async_wait0() {
  asm volatile("s_wait_asynccnt 0x0" ::: "memory");
}

// ---------------- per-channel mean/std thresholds ----------------
__global__ void stats_kernel(const float* __restrict__ x, float* __restrict__ thr) {
  int bc = blockIdx.x;                       // 0..511  (b*128+c)
  const float* p = x + (size_t)bc * HW_PIX;
  float sp = 0.f, sp2 = 0.f, sn = 0.f, sn2 = 0.f;
  for (int i = threadIdx.x; i < HW_PIX; i += blockDim.x) {
    float v = p[i];
    float a = fmaxf(v, 0.f), b = fmaxf(-v, 0.f);
    sp += a; sp2 += a * a; sn += b; sn2 += b * b;
  }
  __shared__ float r0[256], r1[256], r2[256], r3[256];
  int t = threadIdx.x;
  r0[t] = sp; r1[t] = sp2; r2[t] = sn; r3[t] = sn2;
  __syncthreads();
  for (int s = 128; s > 0; s >>= 1) {
    if (t < s) { r0[t] += r0[t+s]; r1[t] += r1[t+s]; r2[t] += r2[t+s]; r3[t] += r3[t+s]; }
    __syncthreads();
  }
  if (t == 0) {
    const float N = (float)HW_PIX;
    float mp = r0[0] / N, mn = r2[0] / N;
    float vp = fmaxf((r1[0] - N * mp * mp) / (N - 1.f), 0.f);   // unbiased (ddof=1)
    float vn = fmaxf((r3[0] - N * mn * mn) / (N - 1.f), 0.f);
    thr[bc]       = mp + 2.0f * sqrtf(vp);
    thr[512 + bc] = mn + 2.0f * sqrtf(vn);
  }
}

// ------------- level-0: down0 = gamma * maxpool3(field(x)) -------------
// channels n: 0..511 pos-field, 512..1023 neg-field (same x channel)
__global__ void down0_kernel(const float* __restrict__ x, const float* __restrict__ thr,
                             float* __restrict__ down0) {
  int n = blockIdx.z;
  int bc = n & 511;
  bool neg = n >= 512;
  float t = thr[n];
  __shared__ float f[18][19];
  int tx = threadIdx.x, ty = threadIdx.y;
  int x0 = blockIdx.x * 16, y0 = blockIdx.y * 16;
  const float* xp = x + (size_t)bc * HW_PIX;
  for (int i = ty * 16 + tx; i < 18 * 18; i += 256) {
    int hy = i / 18, hx = i % 18;
    int gy = y0 + hy - 1, gx = x0 + hx - 1;
    float v = 0.f;                              // field >= 0, so 0-pad == -inf pad
    if ((unsigned)gy < 256u && (unsigned)gx < 256u) {
      float xv = xp[gy * IMG_W + gx];
      float rv = neg ? fmaxf(-xv, 0.f) : fmaxf(xv, 0.f);
      v = fieldval(rv, t);
    }
    f[hy][hx] = v;
  }
  __syncthreads();
  float m = 0.f;
  #pragma unroll
  for (int dy = 0; dy < 3; ++dy)
    #pragma unroll
    for (int dx = 0; dx < 3; ++dx)
      m = fmaxf(m, f[ty + dy][tx + dx]);
  down0[(size_t)n * HW_PIX + (size_t)(y0 + ty) * IMG_W + (x0 + tx)] = GAMMA_F * m;
}

// ------------- pyramid level: dst = scale * maxpool3(bilinear_resize(src, s)) -------------
__global__ void pyr_kernel(const float* __restrict__ src, float* __restrict__ dst,
                           int sin, int s, float scale) {
  int n  = blockIdx.z;
  int xo = blockIdx.x * 16 + threadIdx.x;
  int yo = blockIdx.y * 16 + threadIdx.y;
  if (xo >= s || yo >= s) return;
  const float* sp = src + (size_t)n * sin * sin;
  float r = (float)(sin - 1) / (float)(s - 1);   // align_corners=True
  float m = 0.f;
  #pragma unroll
  for (int dy = -1; dy <= 1; ++dy) {
    int yy = yo + dy; if ((unsigned)yy >= (unsigned)s) continue;
    float fy = yy * r; int i0 = (int)fy; int i1 = (i0 + 1 < sin) ? i0 + 1 : sin - 1;
    float wy = fy - (float)i0;
    #pragma unroll
    for (int dx = -1; dx <= 1; ++dx) {
      int xx = xo + dx; if ((unsigned)xx >= (unsigned)s) continue;
      float fx = xx * r; int j0 = (int)fx; int j1 = (j0 + 1 < sin) ? j0 + 1 : sin - 1;
      float wx = fx - (float)j0;
      float v0 = sp[i0 * sin + j0] * (1.f - wx) + sp[i0 * sin + j1] * wx;
      float v1 = sp[i1 * sin + j0] * (1.f - wx) + sp[i1 * sin + j1] * wx;
      m = fmaxf(m, v0 * (1.f - wy) + v1 * wy);
    }
  }
  dst[(size_t)n * s * s + (size_t)yo * s + xo] = scale * m;
}

// ---------------- WMMA operand gather (wave32, 16-bit A/B layout, ISA 7.12.2) ----------------
// A: lane(0..15)->M=lane, lane(16..31)->M=lane-16 with K offset +8
//    VGPR v: K = 2v (+8 if v>=4) (+8 if upper half-wave), pairs packed.
// B stored N-major (B[k][n] at M[n*ld+k]) -> identical gather with row=n.
__device__ __forceinline__ v16h ld16(const _Float16* M, int ld, int rbase, int kb, int kmax) {
  int lane = threadIdx.x & 31;
  int row  = rbase + (lane & 15);
  int khi  = (lane & 16) ? 8 : 0;
  v16h r;
  #pragma unroll
  for (int v = 0; v < 8; ++v) {
    int k = kb + 2 * v + ((v & 4) ? 8 : 0) + khi;
    r[2 * v]     = (k     < kmax) ? M[row * ld + k]     : (_Float16)0.f;
    r[2 * v + 1] = (k + 1 < kmax) ? M[row * ld + k + 1] : (_Float16)0.f;
  }
  return r;
}

#define RST  136   // radh row stride (halves): bank-conflict-free, 4B aligned
#define W1ST 132
#define HST  18
#define W2ST 18

// ---------------- fused: gather L, radiance, 1x1convs via WMMA, sigmoid, multiply ----------------
// one block = 128 consecutive pixels of one row; 8 waves x 16-pixel WMMA tiles
__global__ void __launch_bounds__(256) final_kernel(
    const float* __restrict__ x, const float* __restrict__ thr,
    const float* __restrict__ down0, const float* __restrict__ pyr,
    const float* __restrict__ w1, const float* __restrict__ b1,
    const float* __restrict__ w2, const float* __restrict__ b2,
    float* __restrict__ out) {
  __shared__ _Float16 radh[128 * RST];     // radiance A-tiles, [pixel][channel]
  __shared__ _Float16 w1s [16  * W1ST];    // w1[o][c]  -> B of GEMM1 (N-major)
  __shared__ _Float16 w2s [128 * W2ST];    // w2[oc][o] -> B of GEMM2 (N-major)
  __shared__ _Float16 hbuf[128 * HST];     // hidden, [pixel][o]
  __shared__ __attribute__((aligned(16))) float thrp_s[128];
  __shared__ __attribute__((aligned(16))) float thrn_s[128];
  __shared__ __attribute__((aligned(16))) float b2s[128];
  __shared__ __attribute__((aligned(16))) float b1s[16];

  int tid = threadIdx.x;
  int xb = (blockIdx.x & 1) * 128;
  int y  = (blockIdx.x >> 1) & 255;
  int b  = blockIdx.x >> 9;

  // --- async-stage small uniforms (verbatim f32 copies) into LDS: 100 x 16B chunks ---
  if (tid < 100) {
    const float* src;
    float* dst;
    if (tid < 32)      { src = thr + b * 128 + tid * 4;              dst = thrp_s + tid * 4; }
    else if (tid < 64) { src = thr + 512 + b * 128 + (tid - 32) * 4; dst = thrn_s + (tid - 32) * 4; }
    else if (tid < 96) { src = b2 + (tid - 64) * 4;                  dst = b2s + (tid - 64) * 4; }
    else               { src = b1 + (tid - 96) * 4;                  dst = b1s + (tid - 96) * 4; }
    async_copy16(src, dst);
  }
  // overlap: weight conversion f32 -> f16 while async copies are in flight
  for (int i = tid; i < 2048; i += 256) {
    w1s[(i >> 7) * W1ST + (i & 127)] = (_Float16)w1[i];
    w2s[(i >> 4) * W2ST + (i & 15)]  = (_Float16)w2[i];
  }
  async_wait0();
  __syncthreads();

  // L_pos/L_neg gather: field(x) vs down0 vs 7 bilinear-upsampled pyramid levels
  for (int i = tid; i < 128 * 128; i += 256) {
    int px = i & 127, c = i >> 7;
    int gx = xb + px;
    size_t pix = (size_t)(b * 128 + c) * HW_PIX + (size_t)y * IMG_W + gx;
    // prefetch next iteration's down0 cachelines (c advances by 2)
    __builtin_prefetch(down0 + pix + 2 * (size_t)HW_PIX, 0, 1);
    __builtin_prefetch(down0 + pix + 514 * (size_t)HW_PIX, 0, 1);
    float v = x[pix];
    float Lp = fmaxf(fieldval(fmaxf(v, 0.f),  thrp_s[c]), down0[pix]);
    float Ln = fmaxf(fieldval(fmaxf(-v, 0.f), thrn_s[c]), down0[(size_t)512 * HW_PIX + pix]);
    size_t off = 0;
    #pragma unroll
    for (int l = 1; l <= 7; ++l) {
      int s = 256 >> l;
      float r  = (float)(s - 1) * (1.0f / 255.0f);   // upsample s -> 256, align_corners
      float fy = y * r;  int i0 = (int)fy; int i1 = (i0 + 1 < s) ? i0 + 1 : s - 1; float wy = fy - (float)i0;
      float fx = gx * r; int j0 = (int)fx; int j1 = (j0 + 1 < s) ? j0 + 1 : s - 1; float wx = fx - (float)j0;
      const float* Pp = pyr + off + (size_t)(b * 128 + c) * s * s;
      const float* Pn = Pp + (size_t)512 * s * s;
      float up = (Pp[i0*s + j0]*(1.f-wx) + Pp[i0*s + j1]*wx)*(1.f-wy)
               + (Pp[i1*s + j0]*(1.f-wx) + Pp[i1*s + j1]*wx)*wy;
      float un = (Pn[i0*s + j0]*(1.f-wx) + Pn[i0*s + j1]*wx)*(1.f-wy)
               + (Pn[i1*s + j0]*(1.f-wx) + Pn[i1*s + j1]*wx)*wy;
      Lp = fmaxf(Lp, up);
      Ln = fmaxf(Ln, un);
      off += (size_t)1024 * s * s;
    }
    radh[px * RST + c] = (_Float16)(Lp - Ln);
  }
  __syncthreads();

  int lane = tid & 31;
  int wid  = tid >> 5;
  int mb   = wid * 16;            // this wave's 16-pixel tile
  int nloc = lane & 15;
  int mo   = (lane & 16) >> 1;    // +8 rows for upper half-wave (C/D layout)

  // GEMM1: h = relu(rad[16xK128] x w1^T[K128xN16] + b1)
  // two independent accumulators -> halves the dependent WMMA->WMMA chain
  v8f acc0 = {0.f, 0.f, 0.f, 0.f, 0.f, 0.f, 0.f, 0.f};
  v8f acc1 = {0.f, 0.f, 0.f, 0.f, 0.f, 0.f, 0.f, 0.f};
  #pragma unroll
  for (int kb = 0; kb < 128; kb += 64) {
    v16h a0 = ld16(radh, RST, mb, kb,      128);
    v16h b0 = ld16(w1s, W1ST, 0,  kb,      128);
    v16h a1 = ld16(radh, RST, mb, kb + 32, 128);
    v16h b1v = ld16(w1s, W1ST, 0, kb + 32, 128);
    acc0 = __builtin_amdgcn_wmma_f32_16x16x32_f16(false, a0, false, b0,  (short)0, acc0, false, false);
    acc1 = __builtin_amdgcn_wmma_f32_16x16x32_f16(false, a1, false, b1v, (short)0, acc1, false, false);
  }
  {
    float bb = b1s[nloc];
    #pragma unroll
    for (int r = 0; r < 8; ++r) {
      float hv = fmaxf(acc0[r] + acc1[r] + bb, 0.f);
      hbuf[(mb + r + mo) * HST + nloc] = (_Float16)hv;
    }
  }
  __syncthreads();

  // GEMM2: mod = sigmoid(h[16xK16(pad32)] x w2^T[K16xN128] + b2); out = x * mod
  // all 8 independent WMMAs issued first so they pipeline back-to-back
  v16h a2 = ld16(hbuf, HST, mb, 0, 16);
  v8f accs[8];
  #pragma unroll
  for (int nb = 0; nb < 8; ++nb) {
    v16h bm2 = ld16(w2s, W2ST, nb * 16, 0, 16);
    v8f z = {0.f, 0.f, 0.f, 0.f, 0.f, 0.f, 0.f, 0.f};
    accs[nb] = __builtin_amdgcn_wmma_f32_16x16x32_f16(false, a2, false, bm2, (short)0, z, false, false);
  }
  #pragma unroll
  for (int nb = 0; nb < 8; ++nb) {
    int   nglob = nb * 16 + nloc;
    float bias  = b2s[nglob];
    #pragma unroll
    for (int r = 0; r < 8; ++r) {
      int m = mb + r + mo;
      float mv = 1.f / (1.f + __expf(-(accs[nb][r] + bias)));
      size_t oidx = (size_t)(b * 128 + nglob) * HW_PIX + (size_t)y * IMG_W + (xb + m);
      out[oidx] = x[oidx] * mv;   // L2-resident scattered stores, coalesced in GL2
    }
  }
}

extern "C" void kernel_launch(void* const* d_in, const int* in_sizes, int n_in,
                              void* d_out, int out_size, void* d_ws, size_t ws_size,
                              hipStream_t stream) {
  const float* x  = (const float*)d_in[0];
  const float* w1 = (const float*)d_in[1];
  const float* b1 = (const float*)d_in[2];
  const float* w2 = (const float*)d_in[3];
  const float* b2 = (const float*)d_in[4];
  float* out = (float*)d_out;

  // workspace layout (floats): thr[1024] | down0[1024*65536] | pyramid levels 1..7
  float* thr   = (float*)d_ws;
  float* down0 = thr + 1024;
  float* pyr   = down0 + (size_t)1024 * HW_PIX;   // total ws ~ 358 MB

  stats_kernel<<<512, 256, 0, stream>>>(x, thr);
  down0_kernel<<<dim3(16, 16, 1024), dim3(16, 16), 0, stream>>>(x, thr, down0);

  const float* src = down0;
  int sin = 256;
  size_t off = 0;
  for (int l = 1; l <= 7; ++l) {
    int s = 256 >> l;                                   // 128,64,32,16,8,4,2
    float scale = (float)pow(0.99, (double)(1 << l));   // gamma^(2^l)
    dim3 grid((s + 15) / 16, (s + 15) / 16, 1024);
    pyr_kernel<<<grid, dim3(16, 16), 0, stream>>>(src, pyr + off, sin, s, scale);
    src = pyr + off;
    off += (size_t)1024 * s * s;
    sin = s;
  }

  final_kernel<<<2048, 256, 0, stream>>>(x, thr, down0, pyr, w1, b1, w2, b2, out);
}
